// SelfAttention_46411416600730
// MI455X (gfx1250) — compile-verified, matrix operand order checked by reading, not run on hardware
//
#include <hip/hip_runtime.h>

#define C_DIM 512
#define SEQ   4096
#define BATCH 2
#define BQ    32
#define BK    64

typedef unsigned int   u32;
typedef unsigned short u16;
typedef __attribute__((ext_vector_type(16))) __bf16 bf16x16;
typedef __attribute__((ext_vector_type(8)))  float  f32x8;

#if defined(__has_builtin)
#if __has_builtin(__builtin_amdgcn_global_load_async_to_lds_b128) && \
    __has_builtin(__builtin_amdgcn_s_wait_asynccnt)
#define USE_ASYNC_LDS 1
#endif
#if __has_builtin(__builtin_amdgcn_cvt_pk_bf16_f32)
#define HAVE_PK_BF16 1
#endif
#endif

#if USE_ASYNC_LDS
typedef int v4i __attribute__((vector_size(16)));
typedef __attribute__((address_space(1))) v4i* gptr_b128;   // global int4*
typedef __attribute__((address_space(3))) v4i* lptr_b128;   // LDS int4*
#endif

union Frag16 { u32 u[8]; bf16x16 v; };

__device__ __forceinline__ u16 f2bf(float x) {
  u32 u = __float_as_uint(x);
  u += 0x7FFFu + ((u >> 16) & 1u);   // round-to-nearest-even
  return (u16)(u >> 16);
}

// two floats -> packed bf16 pair (lo in [15:0]); native v_cvt_pk_bf16_f32 when
// available (RNE, same semantics as the manual path).
__device__ __forceinline__ u32 pack2bf(float lo, float hi) {
#if HAVE_PK_BF16
  return __builtin_bit_cast(u32, __builtin_amdgcn_cvt_pk_bf16_f32(lo, hi));
#else
  return (u32)f2bf(lo) | ((u32)f2bf(hi) << 16);
#endif
}

__device__ __forceinline__ int frag_k(int j, int h) {
  // ISA 16-bit A layout: VGPR0..3 -> K {0..7} (h=0) / {8..15} (h=1)
  //                      VGPR4..7 -> K {16..23} (h=0) / {24..31} (h=1)
  return (j < 4) ? (2 * j + 8 * h) : (16 + 2 * (j - 4) + 8 * h);
}

// A-style fragment: lane(0..15)=row, packed K pairs from a row-major [16][>=32]
// bf16 source. Also valid for B when the B matrix is the transpose of a
// row-major source (lane then indexes the B column).
__device__ __forceinline__ Frag16 load_frag_rowk(const u16* p, int ld) {
  Frag16 f;
  int lane = threadIdx.x & 31, m = lane & 15, h = lane >> 4;
  const u32* row = (const u32*)(p + (size_t)m * ld);
  #pragma unroll
  for (int j = 0; j < 8; ++j) f.u[j] = row[frag_k(j, h) >> 1];
  return f;
}

__device__ __forceinline__ f32x8 wmma_bf16(const Frag16& a, const Frag16& b, f32x8 c) {
  return __builtin_amdgcn_wmma_f32_16x16x32_bf16(false, a.v, false, b.v,
                                                 (short)0, c, false, false);
}

// 16-byte global->LDS copy: async (ASYNCcnt-tracked, no VGPR round trip) when
// the gfx1250 builtin is available, else synchronous fallback.
__device__ __forceinline__ void cp16(const u16* g, u16* l) {
#if USE_ASYNC_LDS
  __builtin_amdgcn_global_load_async_to_lds_b128(
      (gptr_b128)(void*)g, (lptr_b128)(void*)l, 0, 0);
#else
  *(uint4*)l = *(const uint4*)g;
#endif
}

__device__ __forceinline__ void atomicMinF(float* addr, float v) {
  if (v >= 0.f) atomicMin((int*)addr, __float_as_int(v));
  else          atomicMax((unsigned int*)addr, __float_as_uint(v));
}
__device__ __forceinline__ void atomicMaxF(float* addr, float v) {
  if (v >= 0.f) atomicMax((int*)addr, __float_as_int(v));
  else          atomicMin((unsigned int*)addr, __float_as_uint(v));
}

// ---------------------------------------------------------------- minmax ----
__global__ void k_initmm(float* mm) {
  if (threadIdx.x == 0) {
    mm[0] =  __builtin_inff(); mm[1] = -__builtin_inff();
    mm[2] =  __builtin_inff(); mm[3] = -__builtin_inff();
  }
}

__global__ void k_minmax(const float* __restrict__ x, const float* __restrict__ y,
                         float* __restrict__ mm) {
  __shared__ float red[4][256];
  const size_t n = (size_t)BATCH * C_DIM * SEQ;
  int tid = threadIdx.x;
  float xmn = 3.4e38f, xmx = -3.4e38f, ymn = 3.4e38f, ymx = -3.4e38f;
  for (size_t i = (size_t)blockIdx.x * blockDim.x + tid; i < n;
       i += (size_t)gridDim.x * blockDim.x) {
    float xv = x[i], yv = y[i];
    xmn = fminf(xmn, xv); xmx = fmaxf(xmx, xv);
    ymn = fminf(ymn, yv); ymx = fmaxf(ymx, yv);
  }
  red[0][tid] = xmn; red[1][tid] = xmx; red[2][tid] = ymn; red[3][tid] = ymx;
  __syncthreads();
  for (int s = 128; s > 0; s >>= 1) {
    if (tid < s) {
      red[0][tid] = fminf(red[0][tid], red[0][tid + s]);
      red[1][tid] = fmaxf(red[1][tid], red[1][tid + s]);
      red[2][tid] = fminf(red[2][tid], red[2][tid + s]);
      red[3][tid] = fmaxf(red[3][tid], red[3][tid + s]);
    }
    __syncthreads();
  }
  if (tid == 0) {
    atomicMinF(&mm[0], red[0][0]); atomicMaxF(&mm[1], red[1][0]);
    atomicMinF(&mm[2], red[2][0]); atomicMaxF(&mm[3], red[3][0]);
  }
}

// ------------------------------------------------------------- fp32->bf16 ---
__global__ void k_cvt(const float* __restrict__ src, u16* __restrict__ dst, int n2) {
  int i = blockIdx.x * blockDim.x + threadIdx.x;   // pair index
  if (i < n2) {
    float2 v = ((const float2*)src)[i];
    ((u32*)dst)[i] = pack2bf(v.x, v.y);
  }
}

// ------------------------------------------------------ projection GEMMs ----
// proj[m][n] = norm(X[b][:,m]) . W[n][:] + bias[n]
// n <  C_DIM -> OutA[b][m][n]        (row-major, ld C_DIM)   [q / k]
// n >= C_DIM -> OutT[b][n-C_DIM][m]  (transposed, ld SEQ)    [v]
__global__ void k_proj(const float* __restrict__ X, const u16* __restrict__ Wb,
                       const float* __restrict__ bias, const float* __restrict__ mmPair,
                       u16* __restrict__ OutA, u16* __restrict__ OutT) {
  __shared__ u16 As[128 * 32];
  const int m0 = blockIdx.x * 128, n0 = blockIdx.y * 64, b = blockIdx.z;
  const float* Xb = X + (size_t)b * C_DIM * SEQ;
  const float mn = mmPair[0], mx = mmPair[1];
  const float sc = 2.0f / (mx - mn);
  const float sb = -mn * sc - 1.0f;
  const int tid = threadIdx.x, w = tid >> 5;
  const int wm = w & 3, wn = w >> 2;
  const int lane = tid & 31, h = lane >> 4, nl = lane & 15;

  f32x8 acc[2][2] = {};
  for (int kc = 0; kc < C_DIM; kc += 32) {
    // stage normalized/transposed A tile; two K-channels per thread -> b32 LDS
    for (int i = tid; i < 128 * 16; i += 256) {
      int mi = i & 127, kk = (i >> 7) * 2;
      const float* xp = Xb + (size_t)(kc + kk) * SEQ + m0 + mi;
      float v0 = xp[0]   * sc + sb;
      float v1 = xp[SEQ] * sc + sb;
      *(u32*)&As[mi * 32 + kk] = pack2bf(v0, v1);
    }
    __syncthreads();
    const u16* Wr = Wb + (size_t)(n0 + wn * 32) * C_DIM + kc;
    __builtin_prefetch(Wr + 32, 0, 3);          // next K-chunk, near scope
    Frag16 a0 = load_frag_rowk(As + (wm * 32 + 0)  * 32, 32);
    Frag16 a1 = load_frag_rowk(As + (wm * 32 + 16) * 32, 32);
    Frag16 b0 = load_frag_rowk(Wr, C_DIM);      // B = W^T (lane = n)
    Frag16 b1 = load_frag_rowk(Wr + (size_t)16 * C_DIM, C_DIM);
    acc[0][0] = wmma_bf16(a0, b0, acc[0][0]);
    acc[0][1] = wmma_bf16(a0, b1, acc[0][1]);
    acc[1][0] = wmma_bf16(a1, b0, acc[1][0]);
    acc[1][1] = wmma_bf16(a1, b1, acc[1][1]);
    __syncthreads();
  }
  #pragma unroll
  for (int si = 0; si < 2; ++si)
    #pragma unroll
    for (int sj = 0; sj < 2; ++sj) {
      int n = n0 + wn * 32 + sj * 16 + nl;
      float bv = bias[n];
      if (n < C_DIM) {
        u16* Ob = OutA + (size_t)b * SEQ * C_DIM;
        #pragma unroll
        for (int r = 0; r < 8; ++r) {
          int m = m0 + wm * 32 + si * 16 + r + 8 * h;
          Ob[(size_t)m * C_DIM + n] = f2bf(acc[si][sj][r] + bv);
        }
      } else {
        u16* Ot = OutT + (size_t)b * C_DIM * SEQ + (size_t)(n - C_DIM) * SEQ;
        #pragma unroll
        for (int r = 0; r < 8; ++r) {
          int m = m0 + wm * 32 + si * 16 + r + 8 * h;
          Ot[m] = f2bf(acc[si][sj][r] + bv);
        }
      }
    }
}

// ------------------------------------------------------- flash attention ----
// Br=32 queries/block, Bc=64 keys/iter, 8 waves, double-buffered async K/V.
// LDS: Q 32K + 2x K 64K + 2x Vt 64K + S 8K + P 4K + stats  (~300KB of 320KB)
__global__ void k_attn(const u16* __restrict__ Qg, const u16* __restrict__ Kg,
                       const u16* __restrict__ Vtg, const float* __restrict__ mm,
                       u16* __restrict__ AOut) {
  extern __shared__ unsigned char smem[];
  u16*   Qs   = (u16*)smem;                       // [BQ][C_DIM]
  u16*   Ksb  = Qs  + BQ * C_DIM;                 // 2 x [BK][C_DIM]
  u16*   Vtb  = Ksb + 2 * BK * C_DIM;             // 2 x [C_DIM][BK]
  float* Ss   = (float*)(Vtb + 2 * C_DIM * BK);   // [BQ][BK]
  u16*   Ps   = (u16*)(Ss + BQ * BK);             // [BQ][BK]
  float* mrow = (float*)(Ps + BQ * BK);           // [BQ]
  float* lrow = mrow + BQ;
  float* crow = lrow + BQ;

  const int tid = threadIdx.x, w = tid >> 5;
  const int lane = tid & 31, h = lane >> 4, nl = lane & 15;
  const int b  = blockIdx.x / (SEQ / BQ);
  const int qt = blockIdx.x % (SEQ / BQ);
  const int sm = w & 1,  sn = w >> 1;   // score mapping: 2x4 tiles of 16x16
  const int pm = w >> 2, pn = w & 3;    // PV mapping: 2(M) x 4(N/128)

  const u16* Q  = Qg  + ((size_t)b * SEQ + (size_t)qt * BQ) * C_DIM;
  const u16* Kb = Kg  + (size_t)b * SEQ * C_DIM;
  const u16* Vb = Vtg + (size_t)b * C_DIM * SEQ;

  { // Q tile: contiguous rows, plain LDS copy (once)
    const uint4* src = (const uint4*)Q;
    uint4* dst = (uint4*)Qs;
    for (int i = tid; i < BQ * C_DIM / 8; i += 256) dst[i] = src[i];
  }
  if (tid < BQ) { mrow[tid] = -3.4e38f; lrow[tid] = 0.f; }

  // stage tile kt into buffer sel: K rows contiguous, Vt rows strided (ld SEQ)
  auto stage = [&](int kt, int sel) {
    u16* KsD = Ksb + sel * BK * C_DIM;
    u16* VtD = Vtb + sel * C_DIM * BK;
    const u16* ks = Kb + (size_t)kt * BK * C_DIM;
    #pragma unroll
    for (int j = 0; j < (BK * C_DIM / 8) / 256; ++j) {   // 16 chunks
      int i = tid + j * 256;
      cp16(ks + i * 8, KsD + i * 8);
    }
    #pragma unroll
    for (int j = 0; j < (C_DIM * BK / 8) / 256; ++j) {   // 16 chunks
      int i = tid + j * 256;
      int ch = i >> 3, e = i & 7;
      cp16(Vb + (size_t)ch * SEQ + kt * BK + e * 8, VtD + ch * BK + e * 8);
    }
  };

  const int NT = SEQ / BK;
#if USE_ASYNC_LDS
  stage(0, 0);
#endif
  f32x8 O[8] = {};
  for (int kt = 0; kt < NT; ++kt) {
    const int sel = kt & 1;
#if USE_ASYNC_LDS
    __builtin_amdgcn_s_wait_asynccnt(0);   // current tile resident (this wave)
    __syncthreads();                       // all waves done; dead buffer free
    if (kt + 1 < NT) stage(kt + 1, sel ^ 1);
#else
    __syncthreads();
    stage(kt, sel);
    __syncthreads();
#endif
    const u16* Ks  = Ksb + sel * BK * C_DIM;
    const u16* Vts = Vtb + sel * C_DIM * BK;
    { // S = Q . K^T / C      (one 16x16 tile per wave)
      f32x8 s = {};
      for (int kc = 0; kc < C_DIM; kc += 32) {
        Frag16 a   = load_frag_rowk(Qs + (size_t)(sm * 16) * C_DIM + kc, C_DIM);
        Frag16 bfr = load_frag_rowk(Ks + (size_t)(sn * 16) * C_DIM + kc, C_DIM);
        s = wmma_bf16(a, bfr, s);
      }
      const float inv = 1.0f / (float)C_DIM;
      #pragma unroll
      for (int r = 0; r < 8; ++r)
        Ss[(sm * 16 + r + 8 * h) * BK + sn * 16 + nl] = s[r] * inv;
    }
    __syncthreads();
    { // online softmax: 8 threads per q-row, shfl-reduced, packed P stores
      int r  = tid >> 3;
      int c8 = (tid & 7) * 8;
      const float* srow = Ss + r * BK;
      float tm = -3.4e38f;
      #pragma unroll
      for (int j = 0; j < 8; ++j) tm = fmaxf(tm, srow[c8 + j]);
      #pragma unroll
      for (int d = 1; d < 8; d <<= 1) tm = fmaxf(tm, __shfl_xor(tm, d, 32));
      float mo  = mrow[r];
      float mn2 = fmaxf(mo, tm);
      float ps = 0.f;
      u16* prow = Ps + r * BK;
      #pragma unroll
      for (int j = 0; j < 8; j += 2) {
        float e0 = __expf(srow[c8 + j]     - mn2);
        float e1 = __expf(srow[c8 + j + 1] - mn2);
        ps += e0 + e1;
        *(u32*)&prow[c8 + j] = pack2bf(e0, e1);
      }
      #pragma unroll
      for (int d = 1; d < 8; d <<= 1) ps += __shfl_xor(ps, d, 32);
      if ((tid & 7) == 0) {
        float corr = __expf(mo - mn2);
        mrow[r] = mn2; lrow[r] = lrow[r] * corr + ps; crow[r] = corr;
      }
    }
    __syncthreads();
    { // O = O*corr + P.V    (wave: 16 q-rows x 128 channels = 8 frags)
      float cf[8];
      #pragma unroll
      for (int r = 0; r < 8; ++r) cf[r] = crow[pm * 16 + r + 8 * h];
      #pragma unroll
      for (int sub = 0; sub < 8; ++sub)
        #pragma unroll
        for (int r = 0; r < 8; ++r) O[sub][r] *= cf[r];
      for (int kc = 0; kc < BK; kc += 32) {
        Frag16 a = load_frag_rowk(Ps + pm * 16 * BK + kc, BK);
        #pragma unroll
        for (int sub = 0; sub < 8; ++sub) {
          Frag16 bv = load_frag_rowk(Vts + (size_t)(pn * 128 + sub * 16) * BK + kc, BK);
          O[sub] = wmma_bf16(a, bv, O[sub]);
        }
      }
    }
  }
  // epilogue: /= rowsum, denormalize with y min/max, store bf16
  const float ymn = mm[2], ymx = mm[3];
  const float ds = (ymx - ymn) * 0.5f;
  float lr[8];
  #pragma unroll
  for (int r = 0; r < 8; ++r) lr[r] = 1.0f / lrow[pm * 16 + r + 8 * h];
  u16* Orow = AOut + ((size_t)b * SEQ + (size_t)qt * BQ) * C_DIM;
  #pragma unroll
  for (int sub = 0; sub < 8; ++sub) {
    int n = pn * 128 + sub * 16 + nl;
    #pragma unroll
    for (int r = 0; r < 8; ++r) {
      int m = pm * 16 + r + 8 * h;
      float val = (O[sub][r] * lr[r] + 1.0f) * ds + ymn;
      Orow[(size_t)m * C_DIM + n] = f2bf(val);
    }
  }
}

// -------------------------------------- final projection + residual ---------
// out[b][n][m] = tau * (AO[b][m][:] . Wp[n][:] + bp[n]) + x[b][n][m]
__global__ void k_final(const u16* __restrict__ AO, const u16* __restrict__ WPb,
                        const float* __restrict__ bp, const float* __restrict__ X,
                        const float* __restrict__ tau, float* __restrict__ Out) {
  const int m0 = blockIdx.x * 128, n0 = blockIdx.y * 64, b = blockIdx.z;
  const u16* A = AO + (size_t)b * SEQ * C_DIM;
  const int tid = threadIdx.x, w = tid >> 5;
  const int wm = w & 3, wn = w >> 2;
  const int lane = tid & 31, h = lane >> 4, nl = lane & 15;

  f32x8 acc[2][2] = {};
  for (int kc = 0; kc < C_DIM; kc += 32) {
    const u16* Ab = A + (size_t)(m0 + wm * 32) * C_DIM + kc;
    const u16* Wr = WPb + (size_t)(n0 + wn * 32) * C_DIM + kc;
    __builtin_prefetch(Wr + 32, 0, 3);
    Frag16 a0 = load_frag_rowk(Ab, C_DIM);
    Frag16 a1 = load_frag_rowk(Ab + (size_t)16 * C_DIM, C_DIM);
    Frag16 b0 = load_frag_rowk(Wr, C_DIM);
    Frag16 b1 = load_frag_rowk(Wr + (size_t)16 * C_DIM, C_DIM);
    acc[0][0] = wmma_bf16(a0, b0, acc[0][0]);
    acc[0][1] = wmma_bf16(a0, b1, acc[0][1]);
    acc[1][0] = wmma_bf16(a1, b0, acc[1][0]);
    acc[1][1] = wmma_bf16(a1, b1, acc[1][1]);
  }
  const float t = tau[0];
  const float* Xb = X + (size_t)b * C_DIM * SEQ;
  float* Ob = Out + (size_t)b * C_DIM * SEQ;
  #pragma unroll
  for (int si = 0; si < 2; ++si)
    #pragma unroll
    for (int sj = 0; sj < 2; ++sj) {
      int n = n0 + wn * 32 + sj * 16 + nl;
      float bv = bp[n];
      #pragma unroll
      for (int r = 0; r < 8; ++r) {
        int m = m0 + wm * 32 + si * 16 + r + 8 * h;
        size_t idx = (size_t)n * SEQ + m;
        Ob[idx] = t * (acc[si][sj][r] + bv) + Xb[idx];
      }
    }
}

// ---------------------------------------------------------------- launch ----
extern "C" void kernel_launch(void* const* d_in, const int* in_sizes, int n_in,
                              void* d_out, int out_size, void* d_ws, size_t ws_size,
                              hipStream_t stream) {
  (void)in_sizes; (void)n_in; (void)out_size; (void)ws_size;
  const float* y_adapt = (const float*)d_in[0];
  const float* x       = (const float*)d_in[1];
  const float* Wq      = (const float*)d_in[2];
  const float* bq      = (const float*)d_in[3];
  const float* Wkv     = (const float*)d_in[4];
  const float* bkv     = (const float*)d_in[5];
  const float* Wp      = (const float*)d_in[6];
  const float* bp      = (const float*)d_in[7];
  const float* tau     = (const float*)d_in[8];

  char* ws = (char*)d_ws;
  size_t off = 0;
  float* mm = (float*)(ws + off); off += 256;
  u16* WqB  = (u16*)(ws + off);   off += (size_t)C_DIM * C_DIM * 2;
  u16* WkvB = (u16*)(ws + off);   off += (size_t)2 * C_DIM * C_DIM * 2;
  u16* WpB  = (u16*)(ws + off);   off += (size_t)C_DIM * C_DIM * 2;
  u16* Qw   = (u16*)(ws + off);   off += (size_t)BATCH * SEQ * C_DIM * 2;
  u16* Kw   = (u16*)(ws + off);   off += (size_t)BATCH * SEQ * C_DIM * 2;
  u16* Vtw  = (u16*)(ws + off);   off += (size_t)BATCH * C_DIM * SEQ * 2;
  u16* AOw  = (u16*)(ws + off);   off += (size_t)BATCH * SEQ * C_DIM * 2;

  k_initmm<<<1, 64, 0, stream>>>(mm);
  k_minmax<<<512, 256, 0, stream>>>(x, y_adapt, mm);

  k_cvt<<<(C_DIM * C_DIM / 2 + 255) / 256, 256, 0, stream>>>(Wq, WqB, C_DIM * C_DIM / 2);
  k_cvt<<<(C_DIM * C_DIM + 255) / 256, 256, 0, stream>>>(Wkv, WkvB, C_DIM * C_DIM);
  k_cvt<<<(C_DIM * C_DIM / 2 + 255) / 256, 256, 0, stream>>>(Wp, WpB, C_DIM * C_DIM / 2);

  dim3 gq(SEQ / 128, C_DIM / 64, BATCH);
  k_proj<<<gq, 256, 0, stream>>>(x, WqB, bq, mm + 0, Qw, Vtw /*unused*/);
  dim3 gkv(SEQ / 128, (2 * C_DIM) / 64, BATCH);
  k_proj<<<gkv, 256, 0, stream>>>(y_adapt, WkvB, bkv, mm + 2, Kw, Vtw);

  size_t smem = (size_t)BQ * C_DIM * 2          // Q
              + (size_t)2 * BK * C_DIM * 2      // K double buffer
              + (size_t)2 * C_DIM * BK * 2      // Vt double buffer
              + (size_t)BQ * BK * 4             // S f32
              + (size_t)BQ * BK * 2             // P bf16
              + (size_t)3 * BQ * 4;             // m,l,corr
  k_attn<<<BATCH * (SEQ / BQ), 256, smem, stream>>>(Qw, Kw, Vtw, mm, AOw);

  dim3 gf(SEQ / 128, C_DIM / 64, BATCH);
  k_final<<<gf, 256, 0, stream>>>(AOw, WpB, bp, x, tau, (float*)d_out);
}
